// UltraFastLRT_29953101922836
// MI455X (gfx1250) — matrix operand order repaired and checked by reading, not compiled, not run on hardware
//
// UltraFastLRT for MI455X (gfx1250, wave32).
//
// Roofline: per-step token GEMVs dominate (Wq+Wo+fW1+fW2 = 168M weights/step).
// bf16 repack halves streaming vs fp32: 336MB/step -> ~0.72ms ceiling @23.3TB/s,
// cut further by device-side adaptive halting (skip guards on every kernel).
// GEMVs: uint4 bf16 loads (8 cols/thread), split-K partials fused with their
// reduction + LN/gate epilogues via atomic last-block election (5 launches/step).
// Encode GEMMs (M=64) use v_wmma_f32_16x16x32_bf16 with B tiles staged to LDS
// by the Tensor Data Mover (tensor_load_to_lds + s_wait_tensorcnt).
// K/V are hoisted out of the step loop (step-invariant in the reference).

#include <hip/hip_runtime.h>

#define H        4096
#define FFN_DIM  16384
#define QDIM     1024
#define NHEADS   32
#define HEADD    128
#define NSQ      64
#define MAXT     50
#define KCH      128       // TDM staging depth (rows per DMA)

typedef __attribute__((ext_vector_type(16))) __bf16 v16bf;
typedef __attribute__((ext_vector_type(8)))  float  v8f;
typedef __attribute__((ext_vector_type(4)))  unsigned v4u;
typedef __attribute__((ext_vector_type(8)))  int    v8i;
typedef __attribute__((ext_vector_type(4)))  int    v4i;

struct Ctl { float n_steps; int halted; };

__device__ __forceinline__ bool skip_step(const Ctl* c, int t) {
  return c != nullptr && t >= 0 && (c->halted || (float)t >= c->n_steps);
}
__device__ __forceinline__ float sigmoidf_(float x) { return 1.f / (1.f + __expf(-x)); }
__device__ __forceinline__ float bf2f(unsigned short h) {
  union { unsigned u; float f; } v; v.u = (unsigned)h << 16; return v.f;
}

// ---------------- block reductions (blockDim must be power of 2) ----------
__device__ __forceinline__ float block_sum(float v, float* red) {
  const int tid = threadIdx.x;
  red[tid] = v; __syncthreads();
  for (int s = blockDim.x >> 1; s > 0; s >>= 1) {
    if (tid < s) red[tid] += red[tid + s];
    __syncthreads();
  }
  float r = red[0]; __syncthreads(); return r;
}
__device__ __forceinline__ float block_max(float v, float* red) {
  const int tid = threadIdx.x;
  red[tid] = v; __syncthreads();
  for (int s = blockDim.x >> 1; s > 0; s >>= 1) {
    if (tid < s) red[tid] = fmaxf(red[tid], red[tid + s]);
    __syncthreads();
  }
  float r = red[0]; __syncthreads(); return r;
}

// ---------------- misc small kernels --------------------------------------
__global__ void k_zero(int* __restrict__ p, int n) {
  const int i = blockIdx.x * blockDim.x + threadIdx.x;
  if (i < n) p[i] = 0;
}

__global__ void k_cvt_bf16(const float* __restrict__ s, unsigned short* __restrict__ d,
                           long long n) {
  long long i = ((long long)blockIdx.x * blockDim.x + threadIdx.x) * 4;
  if (i >= n) return;
  float4 v = *(const float4*)(s + i);
  unsigned short u0 = __builtin_bit_cast(unsigned short, (__bf16)v.x);
  unsigned short u1 = __builtin_bit_cast(unsigned short, (__bf16)v.y);
  unsigned short u2 = __builtin_bit_cast(unsigned short, (__bf16)v.z);
  unsigned short u3 = __builtin_bit_cast(unsigned short, (__bf16)v.w);
  unsigned long long pk = (unsigned long long)u0 | ((unsigned long long)u1 << 16) |
                          ((unsigned long long)u2 << 32) | ((unsigned long long)u3 << 48);
  *(unsigned long long*)(d + i) = pk;
}

__global__ void k_gather_pe(const int* __restrict__ pieces, const float* __restrict__ emb,
                            float* __restrict__ act) {
  const int r = blockIdx.x;
  const float* e = emb + (size_t)pieces[r] * QDIM;
  float* dst = act + (size_t)r * (2 * QDIM);
  for (int c = threadIdx.x; c < QDIM; c += 256) dst[c] = e[c];
}

// ---------------- M=64 GEMM: TDM -> LDS -> v_wmma_f32_16x16x32_bf16 --------
// C[64,N] = A[64,K] @ W[K,N] + bias.  One wave owns a 16-col stripe of C for
// all 4 M-tiles.  B tiles (KCH x 16 fp32, row stride ldw) are pulled by the
// Tensor Data Mover into a per-wave LDS stage; A rows are contiguous b128
// loads converted in-register to bf16.
__global__ void k_gemm64_wmma(const float* __restrict__ A, int lda,
                              const float* __restrict__ W, int ldw,
                              const float* __restrict__ bias,
                              float* __restrict__ C, int ldc,
                              int K, int N) {
  __shared__ float stageB[8][KCH * 16];     // 8 waves x 8KB = 64KB
  const int lane  = threadIdx.x & 31;
  const int wave  = threadIdx.x >> 5;
  const int ntile = blockIdx.x * (blockDim.x >> 5) + wave;
  const int n0    = ntile * 16;
  if (n0 >= N) return;
  const int half = lane >> 4;
  const int l15  = lane & 15;

  const unsigned lds_off = (unsigned)(size_t)(const void*)&stageB[wave][0];
  const volatile float* bw = &stageB[wave][0];

  // D# group1: data_size=4B, tensor_dim0=ldw, tensor_dim1=K,
  // tile_dim0=16, tile_dim1=KCH, tensor_dim0_stride=ldw (2D tile).
  const unsigned d0g1 = 2u << 16;
  const unsigned d1g1 = ((unsigned)ldw & 0xffffu) << 16;
  const unsigned d2g1 = (((unsigned)ldw >> 16) & 0xffffu) | (((unsigned)K & 0xffffu) << 16);
  const unsigned d3g1 = (((unsigned)K >> 16) & 0xffffu) | (16u << 16);
  const unsigned d4g1 = (unsigned)KCH;
  const unsigned d5g1 = (unsigned)ldw;
  const v8i g1 = {(int)d0g1, (int)d1g1, (int)d2g1, (int)d3g1,
                  (int)d4g1, (int)d5g1, 0, 0};
  const v4i gz4 = {0, 0, 0, 0};
  const v8i gz8 = {0, 0, 0, 0, 0, 0, 0, 0};

  v8f acc[4] = {};
  for (int kk = 0; kk < K; kk += KCH) {
    const unsigned long long ga =
        (unsigned long long)(const void*)(W + (size_t)kk * ldw + n0);
    const v4u g0 = {1u, lds_off, (unsigned)(ga & 0xffffffffu),
                    ((unsigned)((ga >> 32) & 0x1ffffffu)) | (2u << 30)};
    __builtin_amdgcn_tensor_load_to_lds(g0, g1, gz4, gz4, gz8, 0);
    __builtin_amdgcn_s_wait_tensorcnt(0);
    asm volatile("" ::: "memory");

    for (int k2 = 0; k2 < KCH; k2 += 32) {
      v16bf b;   // B 32x16: lane col n0+l15; half 0 -> K {0..7,16..23}, half 1 -> K {8..15,24..31}
#pragma unroll
      for (int j = 0; j < 16; ++j)
        b[j] = (__bf16)bw[(k2 + half * 8 + (j & 7) + (j >> 3) * 16) * 16 + l15];
#pragma unroll
      for (int mt = 0; mt < 4; ++mt) {
        v16bf a;  // A 16x32: half 0 -> K {0..7,16..23}, half 1 -> K {8..15,24..31}
        const float* ap = A + (size_t)(mt * 16 + l15) * lda + (kk + k2) + half * 8;
#pragma unroll
        for (int j = 0; j < 8; ++j) { a[j] = (__bf16)ap[j]; a[8 + j] = (__bf16)ap[16 + j]; }
        acc[mt] = __builtin_amdgcn_wmma_f32_16x16x32_bf16(
            false, a, false, b, (short)0, acc[mt], false, false);
      }
    }
  }
  const float bv = bias ? bias[n0 + l15] : 0.f;
#pragma unroll
  for (int mt = 0; mt < 4; ++mt) {
    const int rbase = mt * 16 + half * 8;   // D: lanes 0-15 rows 0-7, 16-31 rows 8-15
#pragma unroll
    for (int r = 0; r < 8; ++r)
      C[(size_t)(rbase + r) * ldc + (n0 + l15)] = acc[mt][r] + bv;
  }
}

// ---------------- per-row LayerNorm (in place, encode phase) ---------------
__global__ void k_ln_rows(float* __restrict__ x, const float* __restrict__ s,
                          const float* __restrict__ b, int n) {
  __shared__ float red[256];
  float* row = x + (size_t)blockIdx.x * n;
  const int tid = threadIdx.x;
  float sum = 0.f, sq = 0.f;
  for (int i = tid; i < n; i += 256) { float v = row[i]; sum += v; sq += v * v; }
  const float ts = block_sum(sum, red);
  const float tq = block_sum(sq, red);
  const float m = ts / n;
  const float inv = rsqrtf(tq / n - m * m + 1e-6f);
  for (int i = tid; i < n; i += 256) row[i] = (row[i] - m) * inv * s[i] + b[i];
}

__global__ void k_pool_mean(const float* __restrict__ x, float* __restrict__ y,
                            int rows, int n) {
  const int j = blockIdx.x * 256 + threadIdx.x;
  float s = 0.f;
  for (int r = 0; r < rows; ++r) s += x[(size_t)r * n + j];
  y[j] = s * (1.f / rows);
}

// ---------------- fused bf16 GEMV: split-K + reduce + LN/gate epilogue -----
struct GemvFuse {
  const unsigned short* W;   // bf16 weights [K,N]
  const float* x;            // [K]
  const float* bias;         // [N]
  float* out;                // [N] gemv result
  float* part;               // split-K partials [ny,N]
  int K, N, chunk, relu, epi;  // epi: 0 none, 1 LN, 2 LN+gate
  int* cnt_x;                // per column-range counters [nx]
  int* cnt_all;              // [1]
  const float* res;          // LN residual: out2 = LN(res + out)
  const float* lns; const float* lnb;
  float* out2;
  float* token;              // gate state
  const float* gW; const float* gb;
  const float* sW; const float* sb;
  Ctl* ctl; int t;
};

__global__ void k_gemv_fused(GemvFuse p) {
  if (skip_step(p.ctl, p.t)) return;
  __shared__ float xs[512];
  __shared__ float red[256];
  __shared__ int s_last;
  const int tid = threadIdx.x;
  const int j0 = (blockIdx.x * 256 + tid) * 8;   // 8 bf16 columns per thread
  const int i0 = blockIdx.y * p.chunk;
  int i1 = i0 + p.chunk; if (i1 > p.K) i1 = p.K;
  float acc[8] = {0.f, 0.f, 0.f, 0.f, 0.f, 0.f, 0.f, 0.f};
  for (int ib = i0; ib < i1; ib += 512) {
    int nn = i1 - ib; if (nn > 512) nn = 512;
    for (int k = tid; k < nn; k += 256) xs[k] = p.x[ib + k];
    __syncthreads();
    const unsigned short* wr = p.W + (size_t)ib * p.N + j0;
    for (int k = 0; k < nn; ++k) {
      if ((k & 31) == 0)                    // gfx1250 global_prefetch_b8
        __builtin_prefetch((const void*)(wr + (size_t)(k + 48) * p.N), 0, 0);
      const uint4 w = *(const uint4*)(wr + (size_t)k * p.N);
      const float xv = xs[k];
      acc[0] = fmaf(xv, bf2f((unsigned short)(w.x & 0xffffu)), acc[0]);
      acc[1] = fmaf(xv, bf2f((unsigned short)(w.x >> 16)),     acc[1]);
      acc[2] = fmaf(xv, bf2f((unsigned short)(w.y & 0xffffu)), acc[2]);
      acc[3] = fmaf(xv, bf2f((unsigned short)(w.y >> 16)),     acc[3]);
      acc[4] = fmaf(xv, bf2f((unsigned short)(w.z & 0xffffu)), acc[4]);
      acc[5] = fmaf(xv, bf2f((unsigned short)(w.z >> 16)),     acc[5]);
      acc[6] = fmaf(xv, bf2f((unsigned short)(w.w & 0xffffu)), acc[6]);
      acc[7] = fmaf(xv, bf2f((unsigned short)(w.w >> 16)),     acc[7]);
    }
    __syncthreads();
  }
  float* pp = p.part + (size_t)blockIdx.y * p.N + j0;
  *(float4*)(pp)     = make_float4(acc[0], acc[1], acc[2], acc[3]);
  *(float4*)(pp + 4) = make_float4(acc[4], acc[5], acc[6], acc[7]);

  // ---- elect last split-K block for this 2048-column range ----
  __threadfence();
  if (tid == 0)
    s_last = (atomicAdd(&p.cnt_x[blockIdx.x], 1) == (int)gridDim.y - 1) ? 1 : 0;
  __syncthreads();
  if (!s_last) return;
  __threadfence();
  const int base = blockIdx.x * 2048;
  for (int c = tid; c < 2048; c += 256) {
    const int j = base + c;
    float s = p.bias ? p.bias[j] : 0.f;
    for (int y = 0; y < (int)gridDim.y; ++y) s += p.part[(size_t)y * p.N + j];
    if (p.relu) s = fmaxf(s, 0.f);
    p.out[j] = s;
  }
  if (tid == 0) p.cnt_x[blockIdx.x] = 0;
  if (p.epi == 0) return;

  // ---- elect last column-range; run LN (+gate) epilogue in one block ----
  __threadfence();
  if (tid == 0)
    s_last = (atomicAdd(p.cnt_all, 1) == (int)gridDim.x - 1) ? 1 : 0;
  __syncthreads();
  if (!s_last) return;
  if (tid == 0) *p.cnt_all = 0;
  __threadfence();

  const int n = p.N;
  float sum = 0.f, sq = 0.f;
  for (int i = tid; i < n; i += 256) { float v = p.res[i] + p.out[i]; sum += v; sq += v * v; }
  const float ts = block_sum(sum, red);
  const float tq = block_sum(sq, red);
  const float m = ts / n;
  const float inv = rsqrtf(tq / n - m * m + 1e-6f);
  for (int i = tid; i < n; i += 256) {
    float v = p.res[i] + p.out[i];
    p.out2[i] = (v - m) * inv * p.lns[i] + p.lnb[i];
  }
  if (p.epi < 2) return;
  __syncthreads();
  // gate (active is implied: skipped steps never reach here)
  float a = 0.f;
  for (int i = tid; i < H; i += 256)
    a += p.token[i] * p.gW[i] + p.out2[i] * p.gW[H + i];
  const float gdot = block_sum(a, red);
  const float g = sigmoidf_(gdot + p.gb[0]);
  for (int i = tid; i < H; i += 256)
    p.token[i] = g * p.out2[i] + (1.f - g) * p.token[i];
  __syncthreads();
  float a2 = 0.f;
  for (int i = tid; i < H; i += 256) a2 += p.token[i] * p.sW[i];
  const float sdot = block_sum(a2, red);
  if (tid == 0) {
    const float stop = sigmoidf_(sdot + p.sb[0] + 0.1f * (float)p.t);
    if (stop > 0.95f) p.ctl->halted = 1;
  }
}

// ---------------- fp32 GEMV (one-shot: cW1, pW) ----------------------------
__global__ void k_gemv_f32_part(const float* __restrict__ W, const float* __restrict__ x,
                                float* __restrict__ part, int K, int N, int chunk) {
  __shared__ float xs[512];
  const int tid = threadIdx.x;
  const int j = blockIdx.x * 256 + tid;
  const int i0 = blockIdx.y * chunk;
  const int i1 = (i0 + chunk < K) ? (i0 + chunk) : K;
  float acc = 0.f;
  for (int ib = i0; ib < i1; ib += 512) {
    int nn = i1 - ib; if (nn > 512) nn = 512;
    for (int k = tid; k < nn; k += 256) xs[k] = x[ib + k];
    __syncthreads();
    const float* wr = W + (size_t)ib * N + j;
    for (int k = 0; k < nn; ++k) acc = fmaf(xs[k], wr[(size_t)k * N], acc);
    __syncthreads();
  }
  part[(size_t)blockIdx.y * N + j] = acc;
}

__global__ void k_gemv_reduce(const float* __restrict__ part, const float* __restrict__ bias,
                              float* __restrict__ out, int N, int nk, int relu) {
  const int j = blockIdx.x * 256 + threadIdx.x;
  float s = bias ? bias[j] : 0.f;
  for (int k = 0; k < nk; ++k) s += part[(size_t)k * N + j];
  if (relu) s = fmaxf(s, 0.f);
  out[j] = s;
}

// ---------------- complexity head + state init -----------------------------
__global__ void k_init(const float* __restrict__ ctmp, const float* __restrict__ cW2,
                       const float* __restrict__ cb2, const int* __restrict__ max_steps,
                       const float* __restrict__ init_token,
                       float* __restrict__ token, Ctl* ctl) {
  __shared__ float red[256];
  const int tid = threadIdx.x;
  float acc = 0.f;
  for (int i = tid; i < H; i += 256) acc += ctmp[i] * cW2[i];
  const float d = block_sum(acc, red);
  for (int i = tid; i < H; i += 256) token[i] = init_token[i];
  if (tid == 0) {
    const float cpx = sigmoidf_(d + cb2[0]);
    const float ms = (float)max_steps[0];
    ctl->n_steps = 4.f + rintf(cpx * (ms - 4.f));   // MIN_STEPS + round(.)
    ctl->halted = 0;
  }
}

// ---------------- 32-head cross attention over 64 keys ---------------------
__global__ void k_attn(const float* __restrict__ q, const float* __restrict__ Kb,
                       const float* __restrict__ Vb, float* __restrict__ o,
                       const Ctl* ctl, int t) {
  if (skip_step(ctl, t)) return;
  const int h = blockIdx.x;
  const int tid = threadIdx.x;          // blockDim = 128 = HEADD
  __shared__ float aw[NSQ];
  const float* qh = q + h * HEADD;
  if (tid < NSQ) {
    const float* kr = Kb + (size_t)tid * H + h * HEADD;
    float acc = 0.f;
    for (int d = 0; d < HEADD; ++d) acc = fmaf(qh[d], kr[d], acc);
    aw[tid] = acc * 0.088388347648318447f;   // 1/sqrt(128)
  }
  __syncthreads();
  if (tid == 0) {
    float m = aw[0];
    for (int k = 1; k < NSQ; ++k) m = fmaxf(m, aw[k]);
    float s = 0.f;
    for (int k = 0; k < NSQ; ++k) { aw[k] = __expf(aw[k] - m); s += aw[k]; }
    const float inv = 1.f / s;
    for (int k = 0; k < NSQ; ++k) aw[k] *= inv;
  }
  __syncthreads();
  float acc = 0.f;
  for (int k = 0; k < NSQ; ++k)
    acc = fmaf(aw[k], Vb[(size_t)k * H + h * HEADD + tid], acc);
  o[h * HEADD + tid] = acc;
}

// ---------------- value head + policy softmax ------------------------------
__global__ void k_final(const float* __restrict__ token, const float* __restrict__ vW,
                        const float* __restrict__ vb, const float* __restrict__ logits,
                        float* __restrict__ out) {
  __shared__ float red[256];
  const int tid = threadIdx.x;
  float acc = 0.f;
  for (int i = tid; i < H; i += 256) acc += token[i] * vW[i];
  const float v = block_sum(acc, red);
  if (tid == 0) out[0] = 100.f * tanhf(v + vb[0]);
  float mx = -3.4e38f;
  for (int i = tid; i < H; i += 256) mx = fmaxf(mx, logits[i]);
  const float m = block_max(mx, red);
  float es = 0.f;
  for (int i = tid; i < H; i += 256) es += __expf(logits[i] - m);
  const float tot = block_sum(es, red);
  const float inv = 1.f / tot;
  for (int i = tid; i < H; i += 256) out[1 + i] = __expf(logits[i] - m) * inv;
}

// ======================= host: launch sequence =============================
extern "C" void kernel_launch(void* const* d_in, const int* in_sizes, int n_in,
                              void* d_out, int out_size, void* d_ws, size_t ws_size,
                              hipStream_t stream) {
  const int*   pieces = (const int*)  d_in[0];
  const int*   maxst  = (const int*)  d_in[1];
  const float* emb    = (const float*)d_in[2];
  const float* posW   = (const float*)d_in[3];
  const float* posb   = (const float*)d_in[4];
  const float* projW  = (const float*)d_in[5];
  const float* projb  = (const float*)d_in[6];
  const float* ln0s   = (const float*)d_in[7];
  const float* ln0b   = (const float*)d_in[8];
  const float* initT  = (const float*)d_in[9];
  const float* cW1    = (const float*)d_in[10];
  const float* cb1    = (const float*)d_in[11];
  const float* cW2    = (const float*)d_in[12];
  const float* cb2    = (const float*)d_in[13];
  const float* Wq     = (const float*)d_in[14];
  const float* bq     = (const float*)d_in[15];
  const float* Wk     = (const float*)d_in[16];
  const float* bk     = (const float*)d_in[17];
  const float* Wv     = (const float*)d_in[18];
  const float* bv     = (const float*)d_in[19];
  const float* Wo     = (const float*)d_in[20];
  const float* bo     = (const float*)d_in[21];
  const float* ln1s   = (const float*)d_in[22];
  const float* ln1b   = (const float*)d_in[23];
  const float* fW1    = (const float*)d_in[24];
  const float* fb1    = (const float*)d_in[25];
  const float* fW2    = (const float*)d_in[26];
  const float* fb2    = (const float*)d_in[27];
  const float* ln2s   = (const float*)d_in[28];
  const float* ln2b   = (const float*)d_in[29];
  const float* gW     = (const float*)d_in[30];
  const float* gb     = (const float*)d_in[31];
  const float* sW     = (const float*)d_in[32];
  const float* sb     = (const float*)d_in[33];
  const float* vW     = (const float*)d_in[34];
  const float* vb     = (const float*)d_in[35];
  const float* pW     = (const float*)d_in[36];
  const float* pb     = (const float*)d_in[37];
  (void)in_sizes; (void)n_in; (void)out_size; (void)ws_size; // needs ~345MB ws

  char* cur = (char*)d_ws;
  auto carve = [&](size_t bytes) -> char* {
    char* p = cur; cur += (bytes + 255) & ~(size_t)255; return p;
  };
  unsigned short* Wq_bf  = (unsigned short*)carve((size_t)H * H * 2);
  unsigned short* Wo_bf  = (unsigned short*)carve((size_t)H * H * 2);
  unsigned short* fW1_bf = (unsigned short*)carve((size_t)H * FFN_DIM * 2);
  unsigned short* fW2_bf = (unsigned short*)carve((size_t)H * FFN_DIM * 2);
  float* act    = (float*)carve((size_t)NSQ * 2 * QDIM * 4);
  float* board  = (float*)carve((size_t)NSQ * H * 4);
  float* Kb     = (float*)carve((size_t)NSQ * H * 4);
  float* Vb     = (float*)carve((size_t)NSQ * H * 4);
  float* part   = (float*)carve((size_t)64 * FFN_DIM * 4);
  float* token  = (float*)carve(H * 4);
  float* qvec   = (float*)carve(H * 4);
  float* attno  = (float*)carve(H * 4);
  float* atto   = (float*)carve(H * 4);
  float* h1     = (float*)carve(H * 4);
  float* mid    = (float*)carve(FFN_DIM * 4);
  float* ffo    = (float*)carve(H * 4);
  float* newt   = (float*)carve(H * 4);
  float* pooled = (float*)carve(H * 4);
  float* ctmp   = (float*)carve(H * 4);
  float* logits = (float*)carve(H * 4);
  int*   cnts   = (int*)carve(256);     // [0..15]=cnt_x, [16]=cnt_all
  Ctl*   ctl    = (Ctl*)carve(256);

  k_zero<<<1, 64, 0, stream>>>(cnts, 32);

  // ---- one-time bf16 repack of the step-loop weights ----
  k_cvt_bf16<<<(H * H) / 1024, 256, 0, stream>>>(Wq, Wq_bf, (long long)H * H);
  k_cvt_bf16<<<(H * H) / 1024, 256, 0, stream>>>(Wo, Wo_bf, (long long)H * H);
  k_cvt_bf16<<<(int)(((long long)H * FFN_DIM) / 1024), 256, 0, stream>>>(fW1, fW1_bf, (long long)H * FFN_DIM);
  k_cvt_bf16<<<(int)(((long long)H * FFN_DIM) / 1024), 256, 0, stream>>>(fW2, fW2_bf, (long long)H * FFN_DIM);

  // ---- board encoder (TDM + WMMA GEMMs) + step-invariant K/V ----
  k_gather_pe<<<NSQ, 256, 0, stream>>>(pieces, emb, act);
  k_gemm64_wmma<<<QDIM / 128, 256, 0, stream>>>(act, 2 * QDIM, posW, QDIM, posb,
                                                act + QDIM, 2 * QDIM, QDIM, QDIM);
  k_gemm64_wmma<<<H / 128, 256, 0, stream>>>(act, 2 * QDIM, projW, H, projb,
                                             board, H, 2 * QDIM, H);
  k_ln_rows<<<NSQ, 256, 0, stream>>>(board, ln0s, ln0b, H);
  k_gemm64_wmma<<<H / 128, 256, 0, stream>>>(board, H, Wk, H, bk, Kb, H, H, H);
  k_gemm64_wmma<<<H / 128, 256, 0, stream>>>(board, H, Wv, H, bv, Vb, H, H, H);

  // ---- adaptive step budget ----
  k_pool_mean<<<H / 256, 256, 0, stream>>>(board, pooled, NSQ, H);
  k_gemv_f32_part<<<dim3(H / 256, 16), 256, 0, stream>>>(cW1, pooled, part, H, H, H / 16);
  k_gemv_reduce<<<H / 256, 256, 0, stream>>>(part, cb1, ctmp, H, 16, 1);
  k_init<<<1, 256, 0, stream>>>(ctmp, cW2, cb2, maxst, initT, token, ctl);

  // ---- sequential adaptive loop: 5 launches/step, device-side masking ----
  for (int t = 0; t < MAXT; ++t) {
    GemvFuse q{};
    q.W = Wq_bf; q.x = token; q.bias = bq; q.out = qvec; q.part = part;
    q.K = H; q.N = H; q.chunk = 64; q.relu = 0; q.epi = 0;
    q.cnt_x = cnts; q.cnt_all = cnts + 16; q.ctl = ctl; q.t = t;
    k_gemv_fused<<<dim3(2, 64), 256, 0, stream>>>(q);

    k_attn<<<NHEADS, HEADD, 0, stream>>>(qvec, Kb, Vb, attno, ctl, t);

    GemvFuse o{};
    o.W = Wo_bf; o.x = attno; o.bias = bo; o.out = atto; o.part = part;
    o.K = H; o.N = H; o.chunk = 64; o.relu = 0; o.epi = 1;
    o.cnt_x = cnts; o.cnt_all = cnts + 16;
    o.res = token; o.lns = ln1s; o.lnb = ln1b; o.out2 = h1;
    o.ctl = ctl; o.t = t;
    k_gemv_fused<<<dim3(2, 64), 256, 0, stream>>>(o);

    GemvFuse f1{};
    f1.W = fW1_bf; f1.x = h1; f1.bias = fb1; f1.out = mid; f1.part = part;
    f1.K = H; f1.N = FFN_DIM; f1.chunk = 128; f1.relu = 1; f1.epi = 0;
    f1.cnt_x = cnts; f1.cnt_all = cnts + 16; f1.ctl = ctl; f1.t = t;
    k_gemv_fused<<<dim3(8, 32), 256, 0, stream>>>(f1);

    GemvFuse f2{};
    f2.W = fW2_bf; f2.x = mid; f2.bias = fb2; f2.out = ffo; f2.part = part;
    f2.K = FFN_DIM; f2.N = H; f2.chunk = 256; f2.relu = 0; f2.epi = 2;
    f2.cnt_x = cnts; f2.cnt_all = cnts + 16;
    f2.res = h1; f2.lns = ln2s; f2.lnb = ln2b; f2.out2 = newt;
    f2.token = token; f2.gW = gW; f2.gb = gb; f2.sW = sW; f2.sb = sb;
    f2.ctl = ctl; f2.t = t;
    k_gemv_fused<<<dim3(2, 64), 256, 0, stream>>>(f2);
  }

  // ---- heads ----
  k_gemv_f32_part<<<dim3(H / 256, 16), 256, 0, stream>>>(pW, token, part, H, H, H / 16);
  k_gemv_reduce<<<H / 256, 256, 0, stream>>>(part, pb, logits, H, 16, 0);
  k_final<<<1, 256, 0, stream>>>(token, vW, vb, logits, (float*)d_out);
}
